// WordAttention_6176162972189
// MI455X (gfx1250) — matrix operand order repaired and checked by reading, not compiled, not run on hardware
//
#include <hip/hip_runtime.h>

// ---------------------------------------------------------------------------
// WordAttention (HAN word-level) for MI455X / gfx1250, wave32 + WMMA bf16.
// Phase A: fused bidirectional input-gate GEMM  [131072,300] x [300,60]
//          via v_wmma_f32_16x16x32_bf16, weights pre-swizzled in LDS.
// Phase B: sequential GRU scans, one wave32 per (batch, direction) chain.
// Phase C: word attention (tanh MLP + softmax over 64 words + weighted sum).
// ---------------------------------------------------------------------------

typedef __attribute__((ext_vector_type(16))) __bf16          v16bf;
typedef __attribute__((ext_vector_type(8)))  float           v8f;
typedef __attribute__((ext_vector_type(16))) unsigned short  v16us;

#define B_DIM   64
#define S_DIM   32
#define W_DIM   64
#define E_DIM   300
#define H_DIM   10
#define MROWS   (B_DIM * S_DIM * W_DIM)   // 131072 rows (b,s,w)
#define GPAD    64                        // padded gate columns (60 valid)
#define KT      10                        // ceil(300/32) K-steps of 32

__device__ __forceinline__ unsigned short f2bf(float f) {
    unsigned u = __float_as_uint(f);
    u += 0x7FFFu + ((u >> 16) & 1u);          // round to nearest even
    return (unsigned short)(u >> 16);
}

// ---------------------------------------------------------------------------
// Kernel A: gate GEMM.  G[row, g] = text[row,:] . Wcomb[g,:] + bias[g]
//   g in [0,30)  -> forward gates (r,z,n),  g in [30,60) -> reverse gates.
// One wave computes a 16-row x 64-col tile; 8 waves per block -> 128 rows.
// ---------------------------------------------------------------------------
__global__ __launch_bounds__(256)
void gate_gemm_kernel(const float* __restrict__ text,
                      const float* __restrict__ Wih_f, const float* __restrict__ bih_f,
                      const float* __restrict__ Wih_r, const float* __restrict__ bih_r,
                      float* __restrict__ G)
{
    // B fragments pre-swizzled: frag(kt,nt): lane l = column n0+l%16,
    // dword j = packed bf16 {Wcomb[n][e], Wcomb[n][e+1]}, e = k0 + table-order k of element 2j.
    __shared__ __align__(32) unsigned int ldsW[KT * 4 * 32 * 8];   // 40 KB

    const int tid = threadIdx.x;
    for (int idx = tid; idx < KT * 4 * 32 * 8; idx += 256) {
        int j    = idx & 7;
        int lane = (idx >> 3) & 31;
        int nt   = (idx >> 8) & 3;
        int kt   = idx >> 10;
        int s    = 2 * j;
        int kk   = (s < 8) ? ((lane >> 4) * 8 + s) : (16 + (lane >> 4) * 8 + (s - 8));
        int e    = kt * 32 + kk;
        int g0   = nt * 16 + (lane & 15);
        float w0 = 0.f, w1 = 0.f;
        if (e < E_DIM) {
            if (g0 < 30)      w0 = Wih_f[g0 * E_DIM + e];
            else if (g0 < 60) w0 = Wih_r[(g0 - 30) * E_DIM + e];
            if (g0 < 30)      w1 = Wih_f[g0 * E_DIM + e + 1];
            else if (g0 < 60) w1 = Wih_r[(g0 - 30) * E_DIM + e + 1];
        }
        ldsW[idx] = (unsigned)f2bf(w0) | ((unsigned)f2bf(w1) << 16);
    }
    __syncthreads();

    const int wave    = tid >> 5;
    const int lane    = tid & 31;
    const int rowBase = blockIdx.x * 128 + wave * 16;
    const int m       = lane & 15;
    const int kb      = (lane >> 4) * 8;          // K sub-offset per lane half
    const float* __restrict__ Arow = text + (size_t)(rowBase + m) * E_DIM;

    v8f acc[4];
#pragma unroll
    for (int i = 0; i < 4; ++i) acc[i] = 0.0f;

    for (int kt = 0; kt < KT; ++kt) {
        const int k0 = kt * 32;
        v16us au;
        if (kt < KT - 1) {
            const float4* p0 = reinterpret_cast<const float4*>(Arow + k0 + kb);
            const float4* p1 = reinterpret_cast<const float4*>(Arow + k0 + 16 + kb);
            float4 f0 = p0[0], f1 = p0[1], f2 = p1[0], f3 = p1[1];
            au[0]  = f2bf(f0.x); au[1]  = f2bf(f0.y); au[2]  = f2bf(f0.z); au[3]  = f2bf(f0.w);
            au[4]  = f2bf(f1.x); au[5]  = f2bf(f1.y); au[6]  = f2bf(f1.z); au[7]  = f2bf(f1.w);
            au[8]  = f2bf(f2.x); au[9]  = f2bf(f2.y); au[10] = f2bf(f2.z); au[11] = f2bf(f2.w);
            au[12] = f2bf(f3.x); au[13] = f2bf(f3.y); au[14] = f2bf(f3.z); au[15] = f2bf(f3.w);
            __builtin_prefetch(Arow + k0 + 32, 0, 0);   // global_prefetch_b8
        } else {
            // tail: e in [288,320) but only e<300 valid
#pragma unroll
            for (int i = 0; i < 16; ++i) {
                int K = (i < 8) ? (kb + i) : (16 + kb + (i - 8));
                int e = k0 + K;
                au[i] = (e < E_DIM) ? f2bf(Arow[e]) : (unsigned short)0;
            }
        }
        v16bf a = __builtin_bit_cast(v16bf, au);
#pragma unroll
        for (int nt = 0; nt < 4; ++nt) {
            const v16bf* bp =
                reinterpret_cast<const v16bf*>(&ldsW[(((kt * 4 + nt) * 32) + lane) * 8]);
            v16bf bfrag = *bp;
            acc[nt] = __builtin_amdgcn_wmma_f32_16x16x32_bf16(
                false, a, false, bfrag, (short)0, acc[nt], false, false);
        }
    }

    // C layout: VGPR j, lanes 0-15 -> M=j, lanes 16-31 -> M=8+j; N = lane%16
    const int mh = (lane >> 4) * 8;
    const int nl = lane & 15;
#pragma unroll
    for (int nt = 0; nt < 4; ++nt) {
        int g = nt * 16 + nl;
        float bias = 0.f;
        if (g < 30)      bias = bih_f[g];
        else if (g < 60) bias = bih_r[g - 30];
#pragma unroll
        for (int j = 0; j < 8; ++j) {
            int row = rowBase + mh + j;
            G[(size_t)row * GPAD + g] = acc[nt][j] + bias;
        }
    }
}

// ---------------------------------------------------------------------------
// Kernel B: GRU scans. One wave32 per (batch b, direction).
// Lane g<30 owns gate row g of W_hh; lanes 0..9 carry h.  xg loads are
// software-pipelined one step ahead to hide HBM latency off the chain.
// ---------------------------------------------------------------------------
__device__ __forceinline__ float fast_sigmoid(float x) {
    return __builtin_amdgcn_rcpf(1.f + __expf(-x));
}
__device__ __forceinline__ float fast_tanh(float x) {
    float e2 = __expf(-2.f * fabsf(x));
    float t  = (1.f - e2) * __builtin_amdgcn_rcpf(1.f + e2);
    return copysignf(t, x);
}

__global__ __launch_bounds__(32)
void gru_scan_kernel(const float* __restrict__ G,
                     const float* __restrict__ Whh_f, const float* __restrict__ bhh_f,
                     const float* __restrict__ Whh_r, const float* __restrict__ bhh_r,
                     float* __restrict__ OUT)
{
    const int lane = threadIdx.x;
    const int dir  = blockIdx.x & 1;
    const int b    = blockIdx.x >> 1;
    const float* Whh = dir ? Whh_r : Whh_f;
    const float* bhh = dir ? bhh_r : bhh_f;
    const bool  gv   = lane < 30;

    float whh[H_DIM];
#pragma unroll
    for (int k = 0; k < H_DIM; ++k) whh[k] = gv ? Whh[lane * H_DIM + k] : 0.f;
    const float bh   = gv ? bhh[lane] : 0.f;
    const int   gofs = dir * 30;

    float h = 0.f;
    int   sl0 = dir ? (S_DIM - 1) : 0;
    int   row = b * (S_DIM * W_DIM) + sl0 * W_DIM;      // t = 0 -> (w=0, s_vis=0)
    float xg  = G[(size_t)row * GPAD + gofs + lane];

    for (int t = 0; t < W_DIM * S_DIM; ++t) {
        int tn   = (t < W_DIM * S_DIM - 1) ? t + 1 : t;
        int wn   = tn >> 5;
        int svn  = tn & 31;
        int sln  = dir ? (S_DIM - 1 - svn) : svn;
        int rowN = b * (S_DIM * W_DIM) + sln * W_DIM + wn;
        float xgN = G[(size_t)rowN * GPAD + gofs + lane];   // prefetch next step

        // hg[g] = bhh[g] + sum_k Whh[g,k] * h[k]
        float hg = bh;
#pragma unroll
        for (int k = 0; k < H_DIM; ++k)
            hg = fmaf(whh[k], __shfl(h, k, 32), hg);

        float sig = fast_sigmoid(xg + hg);                  // r on 0..9, z on 10..19
        float r   = __shfl(sig, (lane + 12) & 31, 32);      // lane-20 mod 32
        float nv  = fast_tanh(xg + r * hg);                 // n on lanes 20..29
        float z   = __shfl(sig, (lane + 10) & 31, 32);
        float nn  = __shfl(nv,  (lane + 20) & 31, 32);
        float hNew = fmaf(z, h - nn, nn);                   // (1-z)*n + z*h
        h = (lane < H_DIM) ? hNew : h;

        if (lane < H_DIM)
            OUT[(size_t)row * (2 * H_DIM) + dir * H_DIM + lane] = h;

        xg  = xgN;
        row = rowN;
    }
}

// ---------------------------------------------------------------------------
// Kernel C: word attention per (b,s).  64 threads = one per word.
// ---------------------------------------------------------------------------
#define ATT_DIM (2 * H_DIM)   // 20
#define UA_DIM  (4 * H_DIM)   // 40

__global__ __launch_bounds__(64)
void word_attn_kernel(const float* __restrict__ OUT,
                      const float* __restrict__ Wa, const float* __restrict__ ba,
                      const float* __restrict__ Wc, float* __restrict__ out)
{
    __shared__ float sWa[UA_DIM * ATT_DIM];   // 40x20
    __shared__ float sBa[UA_DIM];
    __shared__ float sWc[UA_DIM];
    __shared__ float sLog[W_DIM];
    __shared__ float sExp[W_DIM];
    __shared__ float sAO[W_DIM * ATT_DIM];    // a[w] * o[w][:]

    const int tid = threadIdx.x;
    for (int i = tid; i < UA_DIM * ATT_DIM; i += 64) sWa[i] = Wa[i];
    if (tid < UA_DIM) { sBa[tid] = ba[tid]; sWc[tid] = Wc[tid]; }
    __syncthreads();

    const int bs = blockIdx.x;                         // b*32 + s
    const float* __restrict__ orow = OUT + (size_t)(bs * W_DIM + tid) * ATT_DIM;
    float o[ATT_DIM];
#pragma unroll
    for (int j = 0; j < ATT_DIM; ++j) o[j] = orow[j];

    float logit = 0.f;
    for (int a = 0; a < UA_DIM; ++a) {
        float acc = sBa[a];
#pragma unroll
        for (int j = 0; j < ATT_DIM; ++j) acc = fmaf(sWa[a * ATT_DIM + j], o[j], acc);
        logit = fmaf(sWc[a], fast_tanh(acc), logit);
    }
    sLog[tid] = logit;
    __syncthreads();

    float mx = -3.0e38f;
    for (int w = 0; w < W_DIM; ++w) mx = fmaxf(mx, sLog[w]);
    float e = __expf(logit - mx);
    sExp[tid] = e;
    __syncthreads();

    float den = 0.f;
    for (int w = 0; w < W_DIM; ++w) den += sExp[w];
    float aw = e * __builtin_amdgcn_rcpf(den);

#pragma unroll
    for (int j = 0; j < ATT_DIM; ++j) sAO[tid * ATT_DIM + j] = aw * o[j];
    __syncthreads();

    if (tid < ATT_DIM) {
        float s = 0.f;
        for (int w = 0; w < W_DIM; ++w) s += sAO[w * ATT_DIM + tid];
        out[(size_t)bs * ATT_DIM + tid] = s;
    }
}

// ---------------------------------------------------------------------------
extern "C" void kernel_launch(void* const* d_in, const int* in_sizes, int n_in,
                              void* d_out, int out_size, void* d_ws, size_t ws_size,
                              hipStream_t stream)
{
    (void)in_sizes; (void)n_in; (void)out_size; (void)ws_size;
    const float* text  = (const float*)d_in[0];
    const float* Wih_f = (const float*)d_in[1];
    const float* Whh_f = (const float*)d_in[2];
    const float* bih_f = (const float*)d_in[3];
    const float* bhh_f = (const float*)d_in[4];
    const float* Wih_r = (const float*)d_in[5];
    const float* Whh_r = (const float*)d_in[6];
    const float* bih_r = (const float*)d_in[7];
    const float* bhh_r = (const float*)d_in[8];
    const float* Wa    = (const float*)d_in[9];
    const float* ba    = (const float*)d_in[10];
    const float* Wc    = (const float*)d_in[11];

    float* G   = (float*)d_ws;                      // [131072, 64] f32 padded gates
    float* OUT = G + (size_t)MROWS * GPAD;          // [131072, 20] f32 GRU outputs

    gate_gemm_kernel<<<MROWS / 128, 256, 0, stream>>>(text, Wih_f, bih_f, Wih_r, bih_r, G);
    gru_scan_kernel<<<B_DIM * 2, 32, 0, stream>>>(G, Whh_f, bhh_f, Whh_r, bhh_r, OUT);
    word_attn_kernel<<<B_DIM * S_DIM, 64, 0, stream>>>(OUT, Wa, ba, Wc, (float*)d_out);
}
